// JBF_net_9526237462698
// MI455X (gfx1250) — matrix operand order repaired and checked by reading, not compiled, not run on hardware
//
#include <hip/hip_runtime.h>

typedef __attribute__((ext_vector_type(16))) _Float16 v16h;
typedef __attribute__((ext_vector_type(8)))  _Float16 v8h;
typedef __attribute__((ext_vector_type(8)))  float    v8f;

// ---------------------------------------------------------------------------
// Pack fp32 OIDHW (or IODHW + spatial flip for convT) weights into the CDNA5
// WMMA A-fragment shape (16-bit A 16x32) with this file's contiguous k order, not ISA 7.12.2's:
//   element e of lane L  ->  M = (L&15)+16*ob,
//   K = (L>=16?16:0) + e
// Packed buffer: [kchunk][ob(2)][lane(32)][16 halves]
// ---------------------------------------------------------------------------
__global__ void pack_w_kernel(const float* __restrict__ wraw,
                              _Float16* __restrict__ wpack,
                              int CinTot, int CoutRaw,
                              int kd, int kh, int kw, int tflip)
{
    int cgrp   = CinTot >> 5;
    int ntap   = kd * kh * kw;
    int chunks = cgrp * ntap;
    int total  = chunks * 2 * 32;
    int tid = blockIdx.x * blockDim.x + threadIdx.x;
    if (tid >= total) return;
    int lane = tid & 31;
    int ob   = (tid >> 5) & 1;
    int c    = tid >> 6;
    int tap  = c / cgrp;
    int grp  = c - tap * cgrp;
    int td = tap / (kh * kw);
    int rr = tap - td * kh * kw;
    int th = rr / kw;
    int tw = rr - th * kw;
    int m = (lane & 15) + 16 * ob;
    _Float16* out = wpack + ((size_t)(c * 2 + ob) * 32 + lane) * 16;
    for (int e = 0; e < 16; ++e) {
        int j = e >> 1, hf = e & 1;
        int kl  = ((lane >= 16) ? 16 : 0) + j * 2 + hf;
        int cin = grp * 32 + kl;
        float wv;
        if (!tflip) {
            wv = wraw[((size_t)m * CinTot + cin) * ntap + tap];
        } else {
            int ft = ((kd - 1 - td) * kh + (kh - 1 - th)) * kw + (kw - 1 - tw);
            wv = wraw[((size_t)cin * CoutRaw + m) * ntap + ft];
        }
        out[e] = (_Float16)wv;
    }
}

// ---------------------------------------------------------------------------
// Implicit-GEMM 3D conv, channels-last fp16 activations, 32 out channels.
// Fully specialized on (KD,KH,KW,PAD,CG) so the K-chunk loop unrolls into a
// dense stream of v_wmma_f32_16x16x32_f16. Each wave computes 32 out-channels
// x 16 pixels x 2 rows; weight A-fragments are reused across both rows.
// ---------------------------------------------------------------------------
template <int KD, int KH, int KW, int PAD, int CG>
__global__ __launch_bounds__(32)
void conv_wmma_kernel(const _Float16* __restrict__ src0,
                      const _Float16* __restrict__ src1,
                      const _Float16* __restrict__ wpack,
                      const float* __restrict__ bias,
                      _Float16* __restrict__ dst,
                      int D0full, int d0off, int D1full, int d1off,
                      int Hin, int Win, int Dout, int Hout, int Wout,
                      float slope)
{
    int lane  = threadIdx.x;
    int n     = lane & 15;
    int kbase = (lane >> 4) * 16;
    int w0 = blockIdx.x * 16;
    int h0 = blockIdx.y * 2;           // all layer heights are even
    int bz = blockIdx.z;
    int b  = bz / Dout;
    int d  = bz - b * Dout;
    int w  = w0 + n;
    int wl = (w < Wout) ? w : (Wout - 1);

    v8f acc00, acc10, acc01, acc11;    // [ob][row]
#pragma unroll
    for (int r = 0; r < 8; ++r) {
        acc00[r] = 0.f; acc10[r] = 0.f; acc01[r] = 0.f; acc11[r] = 0.f;
    }

    constexpr int NTAP   = KD * KH * KW;
    constexpr int CHUNKS = NTAP * CG;
#pragma unroll
    for (int c = 0; c < CHUNKS; ++c) {
        const int tap = c / CG;
        const int grp = c - tap * CG;
        const int td = tap / (KH * KW);
        const int rr = tap - td * KH * KW;
        const int th = rr / KW;
        const int tw = rr - th * KW;
        const _Float16* src = grp ? src1 : src0;
        const int Dfull = grp ? D1full : D0full;
        const int doff  = grp ? d1off  : d0off;
        const int id  = d + td + doff;
        const int ih0 = h0 + th - PAD;
        const int ih1 = ih0 + 1;
        const int iw  = wl + tw - PAD;
        const bool wok = (iw >= 0) && (iw < Win);

        v16h b0, b1;
#pragma unroll
        for (int e = 0; e < 16; ++e) { b0[e] = (_Float16)0.0f; b1[e] = (_Float16)0.0f; }
        size_t rowbase = (((size_t)b * Dfull + id) * Hin) * (size_t)Win;
        if (wok && ih0 >= 0 && ih0 < Hin)
            b0 = *(const v16h*)(src + (rowbase + (size_t)ih0 * Win + iw) * 32 + kbase);
        if (wok && ih1 >= 0 && ih1 < Hin)
            b1 = *(const v16h*)(src + (rowbase + (size_t)ih1 * Win + iw) * 32 + kbase);

        const _Float16* wp = wpack + (size_t)(c * 2) * 32 * 16;
        v16h a0 = *(const v16h*)(wp + (size_t)lane * 16);
        v16h a1 = *(const v16h*)(wp + (size_t)(32 + lane) * 16);
        if (c + 1 < CHUNKS)
            __builtin_prefetch(wpack + (size_t)((c + 1) * 2) * 32 * 16, 0, 1);

        acc00 = __builtin_amdgcn_wmma_f32_16x16x32_f16(false, a0, false, b0,
                                                       (short)0, acc00, false, false);
        acc10 = __builtin_amdgcn_wmma_f32_16x16x32_f16(false, a1, false, b0,
                                                       (short)0, acc10, false, false);
        acc01 = __builtin_amdgcn_wmma_f32_16x16x32_f16(false, a0, false, b1,
                                                       (short)0, acc01, false, false);
        acc11 = __builtin_amdgcn_wmma_f32_16x16x32_f16(false, a1, false, b1,
                                                       (short)0, acc11, false, false);
    }

    if (w < Wout) {
        int chbase = (lane >> 4) * 8;   // C/D rows: lanes16-31 hold M=8..15
        float bl0[8], bl1[8];
#pragma unroll
        for (int r = 0; r < 8; ++r) {
            bl0[r] = bias[chbase + r];
            bl1[r] = bias[16 + chbase + r];
        }
#pragma unroll
        for (int row = 0; row < 2; ++row) {
            const v8f& a0r = row ? acc01 : acc00;
            const v8f& a1r = row ? acc11 : acc10;
            size_t pix = (((size_t)b * Dout + d) * Hout + (h0 + row)) * (size_t)Wout + w;
            v8h o0, o1;
#pragma unroll
            for (int r = 0; r < 8; ++r) {
                float v0 = a0r[r] + bl0[r];
                v0 = v0 > 0.f ? v0 : v0 * slope;
                o0[r] = (_Float16)v0;
                float v1 = a1r[r] + bl1[r];
                v1 = v1 > 0.f ? v1 : v1 * slope;
                o1[r] = (_Float16)v1;
            }
            _Float16* dp = dst + pix * 32;
            *(v8h*)(dp + chbase)      = o0;
            *(v8h*)(dp + 16 + chbase) = o1;
        }
    }
}

// ---------------------------------------------------------------------------
// c1: 1 -> 32 channel 3x3x3 VALID conv, fp32 in -> fp16 channels-last out
// ---------------------------------------------------------------------------
__global__ void conv_c1_kernel(const float* __restrict__ x,
                               const float* __restrict__ w,
                               const float* __restrict__ bias,
                               _Float16* __restrict__ dst, int B)
{
    const int Dout = 13, HW = 254;
    int total = B * Dout * HW * HW;
    int tid = blockIdx.x * blockDim.x + threadIdx.x;
    if (tid >= total) return;
    int wp = tid % HW;
    int t  = tid / HW;
    int hp = t % HW; t /= HW;
    int d  = t % Dout;
    int b  = t / Dout;
    float acc[32];
#pragma unroll
    for (int ch = 0; ch < 32; ++ch) acc[ch] = bias[ch];
    for (int td = 0; td < 3; ++td)
        for (int th = 0; th < 3; ++th)
            for (int tw = 0; tw < 3; ++tw) {
                float xin = x[(((size_t)b * 15 + d + td) * 256 + hp + th) * 256 + (wp + tw)];
                int tap = (td * 3 + th) * 3 + tw;
#pragma unroll
                for (int ch = 0; ch < 32; ++ch)
                    acc[ch] += w[ch * 27 + tap] * xin;
            }
    _Float16* o = dst + (size_t)tid * 32;
#pragma unroll
    for (int ch = 0; ch < 32; ++ch) {
        float v = acc[ch];
        v = v > 0.f ? v : v * 0.01f;
        o[ch] = (_Float16)v;
    }
}

// ---------------------------------------------------------------------------
// u1: transposed conv 32 -> 1, kernel 1x3x3, full pad, fp32 out (guide/prior)
// ---------------------------------------------------------------------------
__global__ void convT_u1_kernel(const _Float16* __restrict__ src,
                                const float* __restrict__ wraw,
                                const float* __restrict__ bias,
                                float* __restrict__ dst, int B)
{
    const int D = 7, Hin = 254, Hout = 256;
    int total = B * D * Hout * Hout;
    int tid = blockIdx.x * blockDim.x + threadIdx.x;
    if (tid >= total) return;
    int w = tid % Hout;
    int t = tid / Hout;
    int h = t % Hout; t /= Hout;
    int dz = t % D;
    int b  = t / D;
    float acc = bias[0];
    for (int th = 0; th < 3; ++th)
        for (int tw = 0; tw < 3; ++tw) {
            int ih = h + th - 2, iw = w + tw - 2;
            if (ih < 0 || ih >= Hin || iw < 0 || iw >= Hin) continue;
            const _Float16* p = src + ((((size_t)b * D + dz) * Hin + ih) * Hin + iw) * 32;
            int ft = (2 - th) * 3 + (2 - tw);
#pragma unroll
            for (int ch = 0; ch < 32; ++ch)
                acc += (float)p[ch] * wraw[ch * 9 + ft];
        }
    dst[tid] = acc;
}

// ---------------------------------------------------------------------------
// dk: two 3^3 VALID convs + relu on the 7^3 spatial kernel -> 27 values
// ---------------------------------------------------------------------------
__global__ void dk_kernel(const float* __restrict__ sk,
                          const float* __restrict__ dw1, const float* __restrict__ db1,
                          const float* __restrict__ dw2, const float* __restrict__ db2,
                          float* __restrict__ dk)
{
    if (threadIdx.x != 0 || blockIdx.x != 0) return;
    float r1[125];
    for (int od = 0; od < 5; ++od)
        for (int oh = 0; oh < 5; ++oh)
            for (int ow = 0; ow < 5; ++ow) {
                float s = db1[0];
                for (int dz = 0; dz < 3; ++dz)
                    for (int dy = 0; dy < 3; ++dy)
                        for (int dx = 0; dx < 3; ++dx)
                            s += sk[(od + dz) * 49 + (oh + dy) * 7 + (ow + dx)]
                               * dw1[(dz * 3 + dy) * 3 + dx];
                r1[(od * 5 + oh) * 5 + ow] = s > 0.f ? s : 0.f;
            }
    for (int od = 0; od < 3; ++od)
        for (int oh = 0; oh < 3; ++oh)
            for (int ow = 0; ow < 3; ++ow) {
                float s = db2[0];
                for (int dz = 0; dz < 3; ++dz)
                    for (int dy = 0; dy < 3; ++dy)
                        for (int dx = 0; dx < 3; ++dx)
                            s += r1[(od + dz) * 25 + (oh + dy) * 5 + (ow + dx)]
                               * dw2[(dz * 3 + dy) * 3 + dx];
                dk[(od * 3 + oh) * 3 + ow] = s > 0.f ? s : 0.f;
            }
}

// ---------------------------------------------------------------------------
// Per-pixel bilateral range weights: 7^3 abs-diff patch -> conv(relu) x2,
// wts[27] = dk * r + 1e-10, plus wsum.  guide is fp32 [B][7][256][256].
// ---------------------------------------------------------------------------
__global__ void wts_kernel(const float* __restrict__ guide,
                           const float* __restrict__ rw1, const float* __restrict__ rb1,
                           const float* __restrict__ rw2, const float* __restrict__ rb2,
                           const float* __restrict__ dk,
                           float* __restrict__ wts, int B)
{
    int total = B * 65536;
    int tid = blockIdx.x * blockDim.x + threadIdx.x;
    if (tid >= total) return;
    int w = tid & 255;
    int h = (tid >> 8) & 255;
    int b = tid >> 16;

    float a[343];
    float center = guide[(((size_t)b * 7 + 3) * 256 + h) * 256 + w];
    for (int dz = 0; dz < 7; ++dz)
        for (int dy = 0; dy < 7; ++dy)
            for (int dx = 0; dx < 7; ++dx) {
                int gh = h + dy - 3, gw = w + dx - 3;
                float v = 0.f;
                if (gh >= 0 && gh < 256 && gw >= 0 && gw < 256)
                    v = guide[(((size_t)b * 7 + dz) * 256 + gh) * 256 + gw];
                a[(dz * 7 + dy) * 7 + dx] = fabsf(v - center);
            }
    float r1[125];
    for (int od = 0; od < 5; ++od)
        for (int oh = 0; oh < 5; ++oh)
            for (int ow = 0; ow < 5; ++ow) {
                float s = rb1[0];
                for (int dz = 0; dz < 3; ++dz)
                    for (int dy = 0; dy < 3; ++dy)
                        for (int dx = 0; dx < 3; ++dx)
                            s += a[(od + dz) * 49 + (oh + dy) * 7 + (ow + dx)]
                               * rw1[(dz * 3 + dy) * 3 + dx];
                r1[(od * 5 + oh) * 5 + ow] = s > 0.f ? s : 0.f;
            }
    float* wp = wts + (size_t)tid * 28;
    float wsum = 0.f;
    for (int od = 0; od < 3; ++od)
        for (int oh = 0; oh < 3; ++oh)
            for (int ow = 0; ow < 3; ++ow) {
                float s = rb2[0];
                for (int dz = 0; dz < 3; ++dz)
                    for (int dy = 0; dy < 3; ++dy)
                        for (int dx = 0; dx < 3; ++dx)
                            s += r1[(od + dz) * 25 + (oh + dy) * 5 + (ow + dx)]
                               * rw2[(dz * 3 + dy) * 3 + dx];
                s = s > 0.f ? s : 0.f;
                int j = (od * 3 + oh) * 3 + ow;
                float wv = dk[j] * s + 1e-10f;
                wp[j] = wv;
                wsum += wv;
            }
    wp[27] = wsum;
}

// ---------------------------------------------------------------------------
// Joint bilateral filter: y = relu(sum(wts*patch)/wsum), d = y - inp_c
// ---------------------------------------------------------------------------
__global__ void jbf_y_kernel(const float* __restrict__ lo,
                             const float* __restrict__ mid,
                             const float* __restrict__ hi,
                             unsigned long long bsLo, unsigned long long bsMid,
                             unsigned long long bsHi,
                             const float* __restrict__ inpc, unsigned long long bsInpc,
                             const float* __restrict__ wts,
                             float* __restrict__ ybuf, float* __restrict__ dbuf, int B)
{
    int total = B * 65536;
    int tid = blockIdx.x * blockDim.x + threadIdx.x;
    if (tid >= total) return;
    int w = tid & 255;
    int h = (tid >> 8) & 255;
    int b = tid >> 16;
    const float* wp = wts + (size_t)tid * 28;
    float s = 0.f;
#pragma unroll
    for (int dz = 0; dz < 3; ++dz) {
        const float* pl = (dz == 0) ? (lo + (size_t)b * bsLo)
                        : (dz == 1) ? (mid + (size_t)b * bsMid)
                                    : (hi + (size_t)b * bsHi);
        for (int dy = 0; dy < 3; ++dy)
            for (int dx = 0; dx < 3; ++dx) {
                int hh = h + dy - 1, ww = w + dx - 1;
                if (hh < 0 || hh >= 256 || ww < 0 || ww >= 256) continue;
                s += wp[(dz * 3 + dy) * 3 + dx] * pl[hh * 256 + ww];
            }
    }
    float y = s / wp[27];
    y = y > 0.f ? y : 0.f;
    float d = y - inpc[(size_t)b * bsInpc + h * 256 + w];
    ybuf[tid] = y;
    dbuf[tid] = d;
}

// out = relu(y + (conv3x3(d)+b)*d)
__global__ void jbf_final_kernel(const float* __restrict__ ybuf,
                                 const float* __restrict__ dbuf,
                                 const float* __restrict__ aw,
                                 const float* __restrict__ ab,
                                 float* __restrict__ dst, int B)
{
    int total = B * 65536;
    int tid = blockIdx.x * blockDim.x + threadIdx.x;
    if (tid >= total) return;
    int w = tid & 255;
    int h = (tid >> 8) & 255;
    int b = tid >> 16;
    float c = ab[0];
    for (int dy = 0; dy < 3; ++dy)
        for (int dx = 0; dx < 3; ++dx) {
            int hh = h + dy - 1, ww = w + dx - 1;
            if (hh < 0 || hh >= 256 || ww < 0 || ww >= 256) continue;
            c += aw[dy * 3 + dx] * dbuf[(size_t)b * 65536 + hh * 256 + ww];
        }
    float dd = dbuf[tid];
    float y = ybuf[tid];
    float o = y + c * dd;
    dst[tid] = o > 0.f ? o : 0.f;
}

// ---------------------------------------------------------------------------
extern "C" void kernel_launch(void* const* d_in, const int* in_sizes, int n_in,
                              void* d_out, int out_size, void* d_ws, size_t ws_size,
                              hipStream_t stream)
{
    (void)in_sizes; (void)n_in; (void)out_size; (void)ws_size;
    const int B = 2;
    // input order: x, spat_kernel, then params in dict order
    const float* xf  = (const float*)d_in[0];
    const float* sk  = (const float*)d_in[1];
    const float* c1w = (const float*)d_in[2];  const float* c1b = (const float*)d_in[3];
    const float* c2w = (const float*)d_in[4];  const float* c2b = (const float*)d_in[5];
    const float* c3w = (const float*)d_in[6];  const float* c3b = (const float*)d_in[7];
    const float* c4w = (const float*)d_in[8];  const float* c4b = (const float*)d_in[9];
    const float* p1w = (const float*)d_in[10]; const float* p1b = (const float*)d_in[11];
    const float* p2w = (const float*)d_in[12]; const float* p2b = (const float*)d_in[13];
    const float* p3w = (const float*)d_in[14]; const float* p3b = (const float*)d_in[15];
    const float* u4w = (const float*)d_in[16]; const float* u4b = (const float*)d_in[17];
    const float* u3w = (const float*)d_in[18]; const float* u3b = (const float*)d_in[19];
    const float* u2w = (const float*)d_in[20]; const float* u2b = (const float*)d_in[21];
    const float* u1w = (const float*)d_in[22]; const float* u1b = (const float*)d_in[23];
    const float* rw1 = (const float*)d_in[24]; const float* rb1 = (const float*)d_in[25];
    const float* rw2 = (const float*)d_in[26]; const float* rb2 = (const float*)d_in[27];
    const float* dw1 = (const float*)d_in[28]; const float* db1 = (const float*)d_in[29];
    const float* dw2 = (const float*)d_in[30]; const float* db2 = (const float*)d_in[31];
    const float* a1w = (const float*)d_in[32]; const float* a1b = (const float*)d_in[33];
    const float* a2w = (const float*)d_in[34]; const float* a2b = (const float*)d_in[35];
    const float* a3w = (const float*)d_in[36]; const float* a3b = (const float*)d_in[37];
    const float* a4w = (const float*)d_in[38]; const float* a4b = (const float*)d_in[39];

    // output regions: (y, prior, f1, f2, f3)
    float* outp  = (float*)d_out;
    float* y_out = outp;
    float* prior = outp + 131072;
    float* f1    = outp + 1048576;
    float* f2    = f1 + 131072;
    float* f3    = f2 + 131072;

    // workspace carve-out
    char* ws = (char*)d_ws;
    size_t off = 0;
    auto take = [&](size_t bytes) -> void* {
        void* p = ws + off;
        off += (bytes + 255) & ~(size_t)255;
        return p;
    };
    _Float16* wp_c2 = (_Float16*)take(27 * 2048);
    _Float16* wp_c3 = (_Float16*)take(27 * 2048);
    _Float16* wp_c4 = (_Float16*)take(27 * 2048);
    _Float16* wp_u4 = (_Float16*)take(9 * 2048);
    _Float16* wp_u3 = (_Float16*)take(9 * 2048);
    _Float16* wp_u2 = (_Float16*)take(9 * 2048);
    _Float16* wp_p1 = (_Float16*)take(2 * 2048);
    _Float16* wp_p2 = (_Float16*)take(2 * 2048);
    _Float16* wp_p3 = (_Float16*)take(2 * 2048);
    float* dkbuf = (float*)take(128);
    float* wtsb  = (float*)take((size_t)131072 * 28 * 4);
    float* ybuf  = (float*)take((size_t)131072 * 4);
    float* dbuf  = (float*)take((size_t)131072 * 4);
    _Float16* act1 = (_Float16*)take((size_t)2 * 13 * 254 * 254 * 32 * 2);
    _Float16* act2 = (_Float16*)take((size_t)2 * 11 * 252 * 252 * 32 * 2);
    _Float16* act3 = (_Float16*)take((size_t)2 * 9 * 250 * 250 * 32 * 2);
    _Float16* bufX = (_Float16*)take((size_t)2 * 7 * 254 * 254 * 32 * 2);
    _Float16* bufY = (_Float16*)take((size_t)2 * 7 * 254 * 254 * 32 * 2);

    // ---- pack weights into WMMA A-fragment layout ----
    pack_w_kernel<<<(27 * 64 + 63) / 64, 64, 0, stream>>>(c2w, wp_c2, 32, 32, 3, 3, 3, 0);
    pack_w_kernel<<<(27 * 64 + 63) / 64, 64, 0, stream>>>(c3w, wp_c3, 32, 32, 3, 3, 3, 0);
    pack_w_kernel<<<(27 * 64 + 63) / 64, 64, 0, stream>>>(c4w, wp_c4, 32, 32, 3, 3, 3, 0);
    pack_w_kernel<<<(9 * 64 + 63) / 64, 64, 0, stream>>>(u4w, wp_u4, 32, 32, 1, 3, 3, 1);
    pack_w_kernel<<<(9 * 64 + 63) / 64, 64, 0, stream>>>(u3w, wp_u3, 32, 32, 1, 3, 3, 1);
    pack_w_kernel<<<(9 * 64 + 63) / 64, 64, 0, stream>>>(u2w, wp_u2, 32, 32, 1, 3, 3, 1);
    pack_w_kernel<<<(2 * 64 + 63) / 64, 64, 0, stream>>>(p1w, wp_p1, 64, 32, 1, 1, 1, 0);
    pack_w_kernel<<<(2 * 64 + 63) / 64, 64, 0, stream>>>(p2w, wp_p2, 64, 32, 1, 1, 1, 0);
    pack_w_kernel<<<(2 * 64 + 63) / 64, 64, 0, stream>>>(p3w, wp_p3, 64, 32, 1, 1, 1, 0);

    // ---- prior_est ----
    {
        int tot = B * 13 * 254 * 254;
        conv_c1_kernel<<<(tot + 255) / 256, 256, 0, stream>>>(xf, c1w, c1b, act1, B);
    }

    // conv3: 3x3x3 VALID, 32ch
    auto conv3 = [&](const _Float16* s0, const _Float16* wp, const float* bias,
                     _Float16* dst, int D0, int Hin, int Dout, int Hout) {
        dim3 g((Hout + 15) / 16, Hout / 2, B * Dout);
        conv_wmma_kernel<3, 3, 3, 0, 1><<<g, 32, 0, stream>>>(
            s0, nullptr, wp, bias, dst, D0, 0, 0, 0,
            Hin, Hin, Dout, Hout, Hout, 0.01f);
    };
    // convU: 1x3x3 full pad (transposed conv), 32ch
    auto convU = [&](const _Float16* s0, const _Float16* wp, const float* bias,
                     _Float16* dst, int Hin, int Hout) {
        dim3 g((Hout + 15) / 16, Hout / 2, B * 7);
        conv_wmma_kernel<1, 3, 3, 2, 1><<<g, 32, 0, stream>>>(
            s0, nullptr, wp, bias, dst, 7, 0, 0, 0,
            Hin, Hin, 7, Hout, Hout, 0.01f);
    };
    // convP: 1x1x1 VALID with channel concat (64 -> 32)
    auto convP = [&](const _Float16* s0, const _Float16* s1, const _Float16* wp,
                     const float* bias, _Float16* dst, int D1, int o1, int HW) {
        dim3 g((HW + 15) / 16, HW / 2, B * 7);
        conv_wmma_kernel<1, 1, 1, 0, 2><<<g, 32, 0, stream>>>(
            s0, s1, wp, bias, dst, 7, 0, D1, o1,
            HW, HW, 7, HW, HW, 0.01f);
    };

    conv3(act1, wp_c2, c2b, act2, 13, 254, 11, 252);
    conv3(act2, wp_c3, c3b, act3, 11, 252, 9, 250);
    conv3(act3, wp_c4, c4b, bufX, 9, 250, 7, 248);
    convU(bufX, wp_u4, u4b, bufY, 248, 250);
    convP(bufY, act3, wp_p1, p1b, bufX, 9, 1, 250);
    convU(bufX, wp_u3, u3b, bufY, 250, 252);
    convP(bufY, act2, wp_p2, p2b, bufX, 11, 2, 252);
    convU(bufX, wp_u2, u2b, bufY, 252, 254);
    convP(bufY, act1, wp_p3, p3b, bufX, 13, 3, 254);
    {
        int tot = B * 7 * 256 * 256;
        convT_u1_kernel<<<(tot + 255) / 256, 256, 0, stream>>>(bufX, u1w, u1b, prior, B);
    }

    // ---- bilateral weights ----
    dk_kernel<<<1, 32, 0, stream>>>(sk, dw1, db1, dw2, db2, dkbuf);
    {
        int tot = B * 65536;
        wts_kernel<<<(tot + 127) / 128, 128, 0, stream>>>(prior, rw1, rb1, rw2, rb2,
                                                          dkbuf, wtsb, B);
    }

    // ---- 4 JBF refinement steps ----
    const unsigned long long bsX = 15ull * 65536ull;   // x batch stride (planes)
    const float* loP  = xf + 6 * 65536;
    const float* midP = xf + 7 * 65536;
    const float* hiP  = xf + 8 * 65536;
    const float* inpc = xf + 7 * 65536;
    int tot = B * 65536;
    int blks = (tot + 255) / 256;

    jbf_y_kernel<<<blks, 256, 0, stream>>>(loP, midP, hiP, bsX, bsX, bsX,
                                           inpc, bsX, wtsb, ybuf, dbuf, B);
    jbf_final_kernel<<<blks, 256, 0, stream>>>(ybuf, dbuf, a1w, a1b, f1, B);

    jbf_y_kernel<<<blks, 256, 0, stream>>>(loP, f1, hiP, bsX, 65536ull, bsX,
                                           inpc, bsX, wtsb, ybuf, dbuf, B);
    jbf_final_kernel<<<blks, 256, 0, stream>>>(ybuf, dbuf, a2w, a2b, f2, B);

    jbf_y_kernel<<<blks, 256, 0, stream>>>(loP, f2, hiP, bsX, 65536ull, bsX,
                                           inpc, bsX, wtsb, ybuf, dbuf, B);
    jbf_final_kernel<<<blks, 256, 0, stream>>>(ybuf, dbuf, a3w, a3b, f3, B);

    jbf_y_kernel<<<blks, 256, 0, stream>>>(loP, f3, hiP, bsX, 65536ull, bsX,
                                           inpc, bsX, wtsb, ybuf, dbuf, B);
    jbf_final_kernel<<<blks, 256, 0, stream>>>(ybuf, dbuf, a4w, a4b, y_out, B);
}
